// WindowAttention_19971597926523
// MI455X (gfx1250) — matrix operand order repaired and checked by reading, not compiled, not run on hardware
//
#include <hip/hip_runtime.h>
#include <hip/hip_bf16.h>

// ---------------------------------------------------------------------------
// Fused Swin window-attention for gfx1250 (MI455X), bf16 WMMA + f32 accum.
// One workgroup (8 wave32) per window; all intermediates in LDS.
// Roofline: ~74 GFLOP, ~310 MB mandatory HBM traffic -> both ~13-15 us at
// 23.3 TB/s with bf16 WMMA, hence the fully-fused single-pass structure.
// ---------------------------------------------------------------------------

typedef __attribute__((ext_vector_type(16))) __bf16 v16bf;
typedef __attribute__((ext_vector_type(8)))  float  v8f;

#define NWIN 64
#define NTOK 49
#define NPAD 64
#define CDIM 192
#define NH   6
#define HD   32
#define XS   200   // bf16 row stride for sX/sQ/sK/sO (mult of 8)
#define VTS  72    // bf16 row stride for V^T
#define PS   72    // bf16 row stride for P
#define SS   68    // f32 row stride for S^T ([j][m])

union FragAB { uint4 q[2]; v16bf v; };
union FragC  { v8f v; float f[8]; };

__device__ __forceinline__ unsigned short f2bf(float f) {
  union { float f; unsigned int u; } c; c.f = f;
  unsigned int r = c.u + 0x7FFFu + ((c.u >> 16) & 1u);   // round-to-nearest-even
  return (unsigned short)(r >> 16);
}

// A fragment (16x32 bf16, row-major source): lane L -> row r0+(L&15);
// VGPR v holds K = (v>>2)*16 + (L>>4)*8 + (v&3)*2 (+0,+1)  => two b128 loads.
__device__ __forceinline__ void load_frag_a(FragAB& fr, const unsigned short* base,
                                            int stride, int r0, int k0, int lane) {
  const int half = lane >> 4;
  const unsigned short* p = base + (r0 + (lane & 15)) * stride + k0 + half * 8;
  fr.q[0] = *reinterpret_cast<const uint4*>(p);
  fr.q[1] = *reinterpret_cast<const uint4*>(p + 16);
}

// B fragment (32x16 bf16) loaded from B^T stored row-major (rows = n):
// lane L -> col n0+(L&15); VGPR v holds K = (v>>2)*16 + (L>>4)*8 + (v&3)*2 (+0,+1) => two b128 loads.
__device__ __forceinline__ void load_frag_b(FragAB& fr, const unsigned short* base,
                                            int stride, int n0, int k0, int lane) {
  const int half = lane >> 4;
  const unsigned short* p = base + (n0 + (lane & 15)) * stride + k0 + half * 8;
  fr.q[0] = *reinterpret_cast<const uint4*>(p);
  fr.q[1] = *reinterpret_cast<const uint4*>(p + 16);
}

// ---------------------------------------------------------------------------
// Prep: fp32 weights -> bf16; combined bias+mask table, transposed & padded:
//   comb[w][h][j][m] = rpb[rel[m,j],h] + mask[w,m,j]   (j>=49 -> -1e30)
// ---------------------------------------------------------------------------
__global__ void prep_kernel(const float* __restrict__ qkv_w, const float* __restrict__ proj_w,
                            const float* __restrict__ rpb, const int* __restrict__ rel,
                            const float* __restrict__ mask,
                            unsigned short* __restrict__ wq, unsigned short* __restrict__ wp,
                            float* __restrict__ comb) {
  const int NQ = 3 * CDIM * CDIM;                 // 110592
  const int NP = CDIM * CDIM;                     // 36864
  const int NC = NWIN * NH * NPAD * NPAD;         // 1572864
  for (int i = blockIdx.x * blockDim.x + threadIdx.x; i < NQ + NP + NC;
       i += gridDim.x * blockDim.x) {
    if (i < NQ) {
      wq[i] = f2bf(qkv_w[i]);
    } else if (i < NQ + NP) {
      wp[i - NQ] = f2bf(proj_w[i - NQ]);
    } else {
      int t = i - NQ - NP;
      int m  = t & (NPAD - 1);
      int j  = (t >> 6) & (NPAD - 1);
      int wh = t >> 12;
      int h  = wh % NH;
      int w  = wh / NH;
      float v;
      if (j >= NTOK) {
        v = -1e30f;                               // pad keys killed before softmax
      } else {
        int mm = (m < NTOK) ? m : (NTOK - 1);     // pad rows: finite garbage
        v = rpb[rel[mm * NTOK + j] * NH + h] + mask[w * NTOK * NTOK + mm * NTOK + j];
      }
      comb[t] = v;
    }
  }
}

// ---------------------------------------------------------------------------
// Main fused kernel: one workgroup per window.
// ---------------------------------------------------------------------------
__global__ __launch_bounds__(256)
void win_attn_kernel(const float* __restrict__ x,
                     const float* __restrict__ qkv_b, const float* __restrict__ proj_b,
                     const unsigned short* __restrict__ qkvw,
                     const unsigned short* __restrict__ projw,
                     const float* __restrict__ comb, float* __restrict__ out) {
  __shared__ __attribute__((aligned(16))) unsigned short sX[NPAD * XS];
  __shared__ __attribute__((aligned(16))) unsigned short sQ[NPAD * XS];
  __shared__ __attribute__((aligned(16))) unsigned short sK[NPAD * XS];
  __shared__ __attribute__((aligned(16))) unsigned short sVT[CDIM * VTS]; // V^T [dim][token]
  __shared__ __attribute__((aligned(16))) unsigned short sO[NPAD * XS];
  __shared__ __attribute__((aligned(16))) float          sS[NPAD * SS];   // S^T [j][m]
  __shared__ __attribute__((aligned(16))) unsigned short sP[NPAD * PS];   // P   [m][j]

  const int b    = blockIdx.x;
  const int w    = b & (NWIN - 1);
  const int tid  = threadIdx.x;
  const int wave = tid >> 5;
  const int lane = tid & 31;
  const int half = lane >> 4;
  const float scale = 0.1767766952966369f;        // 32^-0.5

  const int mi = wave & 3;                        // wave's M tile (0..3)
  const int ng = wave >> 2;                       // wave's N group (0..1)
  const int m0 = mi * 16 + half * 8;              // first m of this lane's D column
  const int ln = lane & 15;

  // ---- phase 1: x window -> LDS bf16 (4 threads/row, float4 loads) ---------
  {
    const float* xg = x + (size_t)b * NTOK * CDIM;
    const int m1 = tid >> 2;                      // 0..63
    const int c1 = (tid & 3) * 48;                // 0,48,96,144
    #pragma unroll
    for (int kb = 0; kb < 48; kb += 8) {
      uint4 pk = {0u, 0u, 0u, 0u};
      if (m1 < NTOK) {
        const float4 f0 = *reinterpret_cast<const float4*>(xg + m1 * CDIM + c1 + kb);
        const float4 f1 = *reinterpret_cast<const float4*>(xg + m1 * CDIM + c1 + kb + 4);
        pk.x = (unsigned int)f2bf(f0.x) | ((unsigned int)f2bf(f0.y) << 16);
        pk.y = (unsigned int)f2bf(f0.z) | ((unsigned int)f2bf(f0.w) << 16);
        pk.z = (unsigned int)f2bf(f1.x) | ((unsigned int)f2bf(f1.y) << 16);
        pk.w = (unsigned int)f2bf(f1.z) | ((unsigned int)f2bf(f1.w) << 16);
      }
      *reinterpret_cast<uint4*>(&sX[m1 * XS + c1 + kb]) = pk;
    }
  }
  __syncthreads();

  // ---- phase 2: qkv = x @ W^T + b (three static region loops) --------------
  {
    FragAB aq[6];
    #pragma unroll
    for (int kk = 0; kk < 6; ++kk) load_frag_a(aq[kk], sX, XS, mi * 16, kk * 32, lane);

    // Q region: ni 0..11 -> sQ row-major, pre-scaled by hd^-0.5
    for (int i = 0; i < 6; ++i) {
      const int o0 = (ng + 2 * i) * 16;
      FragC c = {};
      #pragma unroll
      for (int kk = 0; kk < 6; ++kk) {
        FragAB bb;
        load_frag_b(bb, qkvw, CDIM, o0, kk * 32, lane);
        c.v = __builtin_amdgcn_wmma_f32_16x16x32_bf16(false, aq[kk].v, false, bb.v,
                                                      (short)0, c.v, false, false);
      }
      const float bia = qkv_b[o0 + ln];
      #pragma unroll
      for (int r = 0; r < 8; ++r)
        sQ[(m0 + r) * XS + o0 + ln] = f2bf((c.f[r] + bia) * scale);
    }
    // K region: ni 12..23 -> sK row-major
    for (int i = 6; i < 12; ++i) {
      const int o0 = (ng + 2 * i) * 16;
      FragC c = {};
      #pragma unroll
      for (int kk = 0; kk < 6; ++kk) {
        FragAB bb;
        load_frag_b(bb, qkvw, CDIM, o0, kk * 32, lane);
        c.v = __builtin_amdgcn_wmma_f32_16x16x32_bf16(false, aq[kk].v, false, bb.v,
                                                      (short)0, c.v, false, false);
      }
      const float bia = qkv_b[o0 + ln];
      const int oc = o0 - CDIM + ln;
      #pragma unroll
      for (int r = 0; r < 8; ++r)
        sK[(m0 + r) * XS + oc] = f2bf(c.f[r] + bia);
    }
    // V region: ni 24..35 -> sVT transposed, packed b128 store
    for (int i = 12; i < 18; ++i) {
      const int o0 = (ng + 2 * i) * 16;
      FragC c = {};
      #pragma unroll
      for (int kk = 0; kk < 6; ++kk) {
        FragAB bb;
        load_frag_b(bb, qkvw, CDIM, o0, kk * 32, lane);
        c.v = __builtin_amdgcn_wmma_f32_16x16x32_bf16(false, aq[kk].v, false, bb.v,
                                                      (short)0, c.v, false, false);
      }
      const float bia = qkv_b[o0 + ln];
      const int oc = o0 - 2 * CDIM + ln;
      uint4 pk;
      pk.x = (unsigned int)f2bf(c.f[0] + bia) | ((unsigned int)f2bf(c.f[1] + bia) << 16);
      pk.y = (unsigned int)f2bf(c.f[2] + bia) | ((unsigned int)f2bf(c.f[3] + bia) << 16);
      pk.z = (unsigned int)f2bf(c.f[4] + bia) | ((unsigned int)f2bf(c.f[5] + bia) << 16);
      pk.w = (unsigned int)f2bf(c.f[6] + bia) | ((unsigned int)f2bf(c.f[7] + bia) << 16);
      *reinterpret_cast<uint4*>(&sVT[oc * VTS + m0]) = pk;
    }
  }
  __syncthreads();

  // ---- phase 3: per head: S^T = (Q K^T)^T + comb; softmax; O = P V ---------
  for (int h = 0; h < NH; ++h) {
    const float* ch = comb + ((size_t)(w * NH + h)) * NPAD * NPAD;
    {
      FragAB a;
      load_frag_a(a, sQ, XS, mi * 16, h * HD, lane);
      #pragma unroll
      for (int i = 0; i < 2; ++i) {
        const int ni = ng * 2 + i;                // 0..3
        FragAB bb;
        load_frag_b(bb, sK, XS, ni * 16, h * HD, lane);
        FragC c = {};
        c.v = __builtin_amdgcn_wmma_f32_16x16x32_bf16(false, a.v, false, bb.v,
                                                      (short)0, c.v, false, false);
        const int j = ni * 16 + ln;
        // comb is [j][m]: lane's 8 m-values are contiguous -> 2 x b128 loads
        const float4 g0 = *reinterpret_cast<const float4*>(ch + j * NPAD + m0);
        const float4 g1 = *reinterpret_cast<const float4*>(ch + j * NPAD + m0 + 4);
        float4 s0, s1;
        s0.x = c.f[0] + g0.x; s0.y = c.f[1] + g0.y;
        s0.z = c.f[2] + g0.z; s0.w = c.f[3] + g0.w;
        s1.x = c.f[4] + g1.x; s1.y = c.f[5] + g1.y;
        s1.z = c.f[6] + g1.z; s1.w = c.f[7] + g1.w;
        *reinterpret_cast<float4*>(&sS[j * SS + m0])     = s0;  // S^T packed store
        *reinterpret_cast<float4*>(&sS[j * SS + m0 + 4]) = s1;
      }
    }
    __syncthreads();

    // softmax over j for row m (reads S^T: 64 threads -> 64 distinct banks)
    if (tid < NPAD) {
      const int m = tid;
      float mx = -1e30f;
      for (int j = 0; j < NTOK; ++j) mx = fmaxf(mx, sS[j * SS + m]);
      float sum = 0.f;
      for (int j = 0; j < NTOK; ++j) sum += __expf(sS[j * SS + m] - mx);
      const float inv = 1.0f / sum;
      #pragma unroll
      for (int jb = 0; jb < NPAD; jb += 8) {      // P row-contiguous, packed b128
        uint4 pk;
        unsigned int* pu = (unsigned int*)&pk;
        #pragma unroll
        for (int u = 0; u < 4; ++u) {
          const int j0 = jb + 2 * u, j1 = j0 + 1;
          const unsigned short lo =
              (j0 < NTOK) ? f2bf(__expf(sS[j0 * SS + m] - mx) * inv) : (unsigned short)0;
          const unsigned short hi =
              (j1 < NTOK) ? f2bf(__expf(sS[j1 * SS + m] - mx) * inv) : (unsigned short)0;
          pu[u] = (unsigned int)lo | ((unsigned int)hi << 16);
        }
        *reinterpret_cast<uint4*>(&sP[m * PS + jb]) = pk;
      }
    }
    __syncthreads();

    // O tiles: one 16x16 per wave, K = 64 (2 wmma steps); B^T = V^T (stored)
    {
      FragC c = {};
      #pragma unroll
      for (int kk = 0; kk < 2; ++kk) {
        FragAB a, bb;
        load_frag_a(a, sP, PS, mi * 16, kk * 32, lane);
        load_frag_b(bb, sVT, VTS, h * HD + ng * 16, kk * 32, lane);
        c.v = __builtin_amdgcn_wmma_f32_16x16x32_bf16(false, a.v, false, bb.v,
                                                      (short)0, c.v, false, false);
      }
      const int n = h * HD + ng * 16 + ln;
      #pragma unroll
      for (int r = 0; r < 8; ++r)
        sO[(m0 + r) * XS + n] = f2bf(c.f[r]);     // (N,H,hd) layout: col = h*32+d
    }
    __syncthreads();
  }

  // ---- phase 4: out = O @ proj_w^T + proj_b --------------------------------
  {
    FragAB ao[6];
    #pragma unroll
    for (int kk = 0; kk < 6; ++kk) load_frag_a(ao[kk], sO, XS, mi * 16, kk * 32, lane);

    float* og = out + (size_t)b * NTOK * CDIM;
    for (int i = 0; i < 6; ++i) {
      const int ni = ng + 2 * i;                  // 0..11, disjoint across waves
      FragC c = {};
      #pragma unroll
      for (int kk = 0; kk < 6; ++kk) {
        FragAB bb;
        load_frag_b(bb, projw, CDIM, ni * 16, kk * 32, lane);
        c.v = __builtin_amdgcn_wmma_f32_16x16x32_bf16(false, ao[kk].v, false, bb.v,
                                                      (short)0, c.v, false, false);
      }
      const int n = ni * 16 + ln;
      const float bia = proj_b[n];
      #pragma unroll
      for (int r = 0; r < 8; ++r) {
        const int m = m0 + r;
        if (m < NTOK) og[m * CDIM + n] = c.f[r] + bia;
      }
    }
  }
}

// ---------------------------------------------------------------------------
extern "C" void kernel_launch(void* const* d_in, const int* in_sizes, int n_in,
                              void* d_out, int out_size, void* d_ws, size_t ws_size,
                              hipStream_t stream) {
  const float* x      = (const float*)d_in[0];
  const float* mask   = (const float*)d_in[1];
  const float* qkv_w  = (const float*)d_in[2];
  const float* qkv_b  = (const float*)d_in[3];
  const float* proj_w = (const float*)d_in[4];
  const float* proj_b = (const float*)d_in[5];
  const float* rpb    = (const float*)d_in[6];
  const int*   rel    = (const int*)d_in[7];

  // ws: [qkv_w bf16][proj_w bf16][comb f32 (64,6,64,64)]
  unsigned short* wq = (unsigned short*)d_ws;
  unsigned short* wp = wq + 3 * CDIM * CDIM;
  float* comb = (float*)((char*)d_ws + (size_t)(3 * CDIM * CDIM + CDIM * CDIM) * 2);

  prep_kernel<<<1024, 256, 0, stream>>>(qkv_w, proj_w, rpb, rel, mask, wq, wp, comb);
  win_attn_kernel<<<4096, 256, 0, stream>>>(x, qkv_b, proj_b, wq, wp, comb,
                                            (float*)d_out);
}
